// Model_36532991819839
// MI455X (gfx1250) — matrix operand order repaired and checked by reading, not compiled, not run on hardware
//
#include <hip/hip_runtime.h>
#include <hip/hip_bf16.h>

typedef __attribute__((ext_vector_type(16))) __bf16 v16bf;
typedef __attribute__((ext_vector_type(8)))  __bf16 v8bf;
typedef __attribute__((ext_vector_type(8)))  float  v8f;
typedef int v4i __attribute__((__vector_size__(4 * sizeof(int))));
typedef unsigned int u32x4 __attribute__((ext_vector_type(4)));
typedef int i32x4 __attribute__((ext_vector_type(4)));
typedef int i32x8 __attribute__((ext_vector_type(8)));

#define BDIM  16384
#define SEQ   50
#define NFEAT 109
#define DFEAT 128
#define HDIM  512
#define HID   2048
#define NBLK  10
#define NOUT  20

#if __has_builtin(__builtin_amdgcn_global_load_async_to_lds_b128)
#define HAS_ASYNC_LDS 1
#define GLOBAL_TO_LDS_ASYNC_B128(gp, lp)                                   \
  __builtin_amdgcn_global_load_async_to_lds_b128(                          \
      (__attribute__((address_space(1))) v4i*)(gp),                        \
      (__attribute__((address_space(3))) v4i*)(lp), 0, 0)
#else
#define HAS_ASYNC_LDS 0
#endif

#if __has_builtin(__builtin_amdgcn_tensor_load_to_lds)
#define HAS_TDM 1
#if __clang_major__ >= 23
#define TDM_LOAD(g0, g1) \
  __builtin_amdgcn_tensor_load_to_lds(g0, g1, (i32x4)0, (i32x4)0, (i32x8)0, 0)
#else
#define TDM_LOAD(g0, g1) \
  __builtin_amdgcn_tensor_load_to_lds(g0, g1, (i32x4)0, (i32x4)0, 0)
#endif
#else
#define HAS_TDM 0
#endif

#if __has_builtin(__builtin_amdgcn_s_wait_asynccnt)
#define WAIT_ASYNC(n) __builtin_amdgcn_s_wait_asynccnt(n)
#else
#define WAIT_ASYNC(n) asm volatile("s_wait_asynccnt %0" ::"i"(n) : "memory")
#endif

#if __has_builtin(__builtin_amdgcn_s_wait_tensorcnt)
#define WAIT_TENSOR(n) __builtin_amdgcn_s_wait_tensorcnt(n)
#else
#define WAIT_TENSOR(n) asm volatile("s_wait_tensorcnt %0" ::"i"(n) : "memory")
#endif

// ---------------------------------------------------------------------------
// Weight convert + transpose: in is R x C fp32 (row-major), out is C x R bf16.
// ---------------------------------------------------------------------------
__global__ __launch_bounds__(256)
void k_transpose_cvt(const float* __restrict__ in, __bf16* __restrict__ out,
                     int R, int C) {
  __shared__ __bf16 tile[32][33];
  const size_t msz = (size_t)R * C;
  in  += (size_t)blockIdx.z * msz;
  out += (size_t)blockIdx.z * msz;
  const int c0 = blockIdx.x * 32, r0 = blockIdx.y * 32;
  const int tx = threadIdx.x & 31, ty = threadIdx.x >> 5;  // 32 x 8
  #pragma unroll
  for (int i = 0; i < 32; i += 8)
    tile[ty + i][tx] = (__bf16)in[(size_t)(r0 + ty + i) * C + (c0 + tx)];
  __syncthreads();
  #pragma unroll
  for (int i = 0; i < 32; i += 8)
    out[(size_t)(c0 + ty + i) * R + (r0 + tx)] = tile[tx][ty + i];
}

// ---------------------------------------------------------------------------
// Pooling: per batch row compute [avg | clip(max) | clip(min) | next_item].
// ---------------------------------------------------------------------------
__global__ __launch_bounds__(DFEAT)
void k_pool(const float* __restrict__ input, const int* __restrict__ mask,
            const float* __restrict__ embed,
            float* __restrict__ hf, __bf16* __restrict__ hb) {
  const int b = blockIdx.x;
  const int f = threadIdx.x;             // 0..127
  const float* row = input + (size_t)b * SEQ * NFEAT;
  const int* mrow = mask + (size_t)b * SEQ;

  int sl = 0;
  #pragma unroll 5
  for (int s = 0; s < SEQ; ++s) sl += mrow[s];
  if (sl < 1) sl = 1;

  float sum = 0.0f, mx = -3.0e38f, mn = 3.0e38f, last = 0.0f;
  for (int s = 0; s < SEQ; ++s) {
    const float* p = row + s * NFEAT;
    float v;
    if (f < 20) {
      const int id = (int)p[0];
      v = embed[(size_t)id * 20 + f];
    } else {
      v = p[1 + (f - 20)];
    }
    if (s < sl) {
      sum += v;
      mx = fmaxf(mx, v);
      mn = fminf(mn, v);
    }
    if (s == SEQ - 1) last = v;
  }
  const float avg = sum / (float)sl;
  mx = fminf(fmaxf(mx, 1e-9f), 1e9f);
  mn = fminf(fmaxf(mn, 1e-9f), 1e9f);

  const size_t o = (size_t)b * HDIM;
  hf[o + f]       = avg;  hb[o + f]       = (__bf16)avg;
  hf[o + 128 + f] = mx;   hb[o + 128 + f] = (__bf16)mx;
  hf[o + 256 + f] = mn;   hb[o + 256 + f] = (__bf16)mn;
  hf[o + 384 + f] = last; hb[o + 384 + f] = (__bf16)last;
}

#if HAS_TDM
// ---------------------------------------------------------------------------
// TDM descriptor (D#) for a dense 2D tile load: 128 rows x 64 bf16, row
// stride = kstride elements, packed contiguously into LDS at lds_byte_off.
// Groups per ISA 8.3/8.4; groups 2/3 zero (tile <= 2D).
// ---------------------------------------------------------------------------
__device__ __forceinline__ void tdm_tile_load_128x64(const __bf16* gsrc,
                                                     unsigned lds_byte_off,
                                                     long kstride) {
  const unsigned long long ga = (unsigned long long)(size_t)gsrc;
  u32x4 g0;
  g0[0] = 1u;                                            // count=1 (valid D#)
  g0[1] = lds_byte_off;                                  // lds_addr [63:32]
  g0[2] = (unsigned)ga;                                  // global_addr lo
  g0[3] = (unsigned)((ga >> 32) & 0x1FFFFFFu)            // global_addr [56:32]
        | (2u << 30);                                    // type=2 ("image")
  i32x8 g1;
  g1[0] = 1 << 16;                                       // data_size=1 -> 2B
  g1[1] = (int)((kstride & 0xFFFF) << 16);               // tensor_dim0 lo16
  g1[2] = (int)(((kstride >> 16) & 0xFFFF)               // tensor_dim0 hi16
        | (128u << 16));                                 // tensor_dim1 lo16
  g1[3] = 64 << 16;                                      // tile_dim0 = 64
  g1[4] = 128;                                           // tile_dim1 = 128
  g1[5] = (int)kstride;                                  // tensor_dim0_stride
  g1[6] = 0;
  g1[7] = 0;
  TDM_LOAD(g0, g1);
}

__device__ __forceinline__ unsigned lds_off_u32(const void* p) {
  return (unsigned)(size_t)(__attribute__((address_space(3))) const void*)p;
}
#endif

// ---------------------------------------------------------------------------
// Tiled bf16 WMMA GEMM: C[M x N] = act( A[M x K] * BT[N x K]^T + bias )
//   mode 0: out = relu(.) -> bf16 only            (h @ W1 -> t)
//   mode 1: out = resid + relu(.) -> f32 + bf16   (t @ W2 + h -> h)
// 256 threads = 8 waves; 128x128 tile; each wave owns 64x32 (4x2 WMMA tiles).
// BK = 64. LDS double-buffered; tiles staged by the Tensor Data Mover
// (one tensor_load_to_lds per tile, TENSORcnt) when available, else the
// async copy engine (ASYNCcnt), so the next stage overlaps current WMMAs.
// ---------------------------------------------------------------------------
__global__ __launch_bounds__(256)
void k_gemm(const __bf16* __restrict__ A, const __bf16* __restrict__ BT,
            const float* __restrict__ bias, const float* __restrict__ resid,
            float* __restrict__ outF, __bf16* __restrict__ outB,
            int N, int K, int mode) {
#if HAS_TDM || HAS_ASYNC_LDS
  __shared__ __align__(128) __bf16 lsA[2][128][64];
  __shared__ __align__(128) __bf16 lsB[2][128][64];
#else
  __shared__ __align__(128) __bf16 lsA[1][128][64];
  __shared__ __align__(128) __bf16 lsB[1][128][64];
#endif

  const int tid  = threadIdx.x;
  const int lane = tid & 31;
  const int wid  = tid >> 5;
  const int wm   = wid >> 2;        // 0..1  -> 64-row slab
  const int wn   = wid & 3;         // 0..3  -> 32-col slab
  const int ml   = lane & 15;
  const int kh   = lane >> 4;       // K-half select per ISA operand layout
  const long blockM = (long)blockIdx.y * 128;
  const long blockN = (long)blockIdx.x * 128;

  v8f acc[4][2];
  #pragma unroll
  for (int mt = 0; mt < 4; ++mt)
    #pragma unroll
    for (int nt = 0; nt < 2; ++nt)
      #pragma unroll
      for (int e = 0; e < 8; ++e) acc[mt][nt][e] = 0.0f;

  // One 16x16x32 MAC sweep (8 WMMAs) from LDS buffer `buf`, k-slice kk.
  auto mac_step = [&](int buf, int kk) {
    v16bf af[4], bfr[2];
    #pragma unroll
    for (int mt = 0; mt < 4; ++mt) {
      const int r = wm * 64 + mt * 16 + ml;
      const v8bf lo = *(const v8bf*)&lsA[buf][r][kk + kh * 8];
      const v8bf hi = *(const v8bf*)&lsA[buf][r][kk + 16 + kh * 8];
      #pragma unroll
      for (int e = 0; e < 8; ++e) { af[mt][e] = lo[e]; af[mt][8 + e] = hi[e]; }
    }
    #pragma unroll
    for (int nt = 0; nt < 2; ++nt) {
      const int r = wn * 32 + nt * 16 + ml;
      const v8bf lo = *(const v8bf*)&lsB[buf][r][kk + kh * 8];
      const v8bf hi = *(const v8bf*)&lsB[buf][r][kk + 16 + kh * 8];
      #pragma unroll
      for (int e = 0; e < 8; ++e) { bfr[nt][e] = lo[e]; bfr[nt][8 + e] = hi[e]; }
    }
    #pragma unroll
    for (int mt = 0; mt < 4; ++mt)
      #pragma unroll
      for (int nt = 0; nt < 2; ++nt)
        acc[mt][nt] = __builtin_amdgcn_wmma_f32_16x16x32_bf16(
            false, af[mt], false, bfr[nt], (short)0, acc[mt][nt],
            false, false);
  };

#if HAS_TDM
  // --- Tensor Data Mover staging: 2 TDM ops per stage, issued by wave 0. ---
  auto stage = [&](int buf, int k0) {
    if (wid == 0) {
      tdm_tile_load_128x64(A + blockM * (long)K + k0,
                           lds_off_u32(&lsA[buf][0][0]), (long)K);
      tdm_tile_load_128x64(BT + blockN * (long)K + k0,
                           lds_off_u32(&lsB[buf][0][0]), (long)K);
    }
  };
  stage(0, 0);
  int buf = 0;
  for (int k0 = 0; k0 < K; k0 += 64) {
    const bool pf = (k0 + 64) < K;
    if (pf) {
      stage(buf ^ 1, k0 + 64);       // wave0: 4 outstanding tensor ops
      WAIT_TENSOR(2);                // retire current stage (in-order)
    } else {
      WAIT_TENSOR(0);
    }
    __syncthreads();                 // fills visible to all waves
    mac_step(buf, 0);
    mac_step(buf, 32);
    __syncthreads();                 // done reading before next overwrite
    buf ^= 1;
  }
#elif HAS_ASYNC_LDS
  __bf16* Anc  = const_cast<__bf16*>(A);
  __bf16* BTnc = const_cast<__bf16*>(BT);
  auto stage = [&](int buf, int k0) {
    #pragma unroll
    for (int it = 0; it < 4; ++it) {
      const int c  = tid + it * 256;
      const int r  = c >> 3;
      const int ch = (c & 7) * 8;
      GLOBAL_TO_LDS_ASYNC_B128(Anc + (blockM + r) * (long)K + (k0 + ch),
                               &lsA[buf][r][ch]);
    }
    #pragma unroll
    for (int it = 0; it < 4; ++it) {
      const int c  = tid + it * 256;
      const int r  = c >> 3;
      const int ch = (c & 7) * 8;
      GLOBAL_TO_LDS_ASYNC_B128(BTnc + (blockN + r) * (long)K + (k0 + ch),
                               &lsB[buf][r][ch]);
    }
  };
  stage(0, 0);
  int buf = 0;
  for (int k0 = 0; k0 < K; k0 += 64) {
    const bool pf = (k0 + 64) < K;
    if (pf) {
      stage(buf ^ 1, k0 + 64);
      WAIT_ASYNC(8);
    } else {
      WAIT_ASYNC(0);
    }
    __syncthreads();
    mac_step(buf, 0);
    mac_step(buf, 32);
    __syncthreads();
    buf ^= 1;
  }
#else
  for (int k0 = 0; k0 < K; k0 += 64) {
    #pragma unroll
    for (int it = 0; it < 4; ++it) {
      const int c  = tid + it * 256;
      const int r  = c >> 3;
      const int ch = (c & 7) * 8;
      *(uint4*)&lsA[0][r][ch] =
          *(const uint4*)(A + (blockM + r) * (long)K + (k0 + ch));
    }
    #pragma unroll
    for (int it = 0; it < 4; ++it) {
      const int c  = tid + it * 256;
      const int r  = c >> 3;
      const int ch = (c & 7) * 8;
      *(uint4*)&lsB[0][r][ch] =
          *(const uint4*)(BT + (blockN + r) * (long)K + (k0 + ch));
    }
    __syncthreads();
    mac_step(0, 0);
    mac_step(0, 32);
    __syncthreads();
  }
#endif

  // Epilogue: bias + ReLU (+ residual for mode 1).
  #pragma unroll
  for (int nt = 0; nt < 2; ++nt) {
    const long gn = blockN + wn * 32 + nt * 16 + ml;
    const float bb = bias[gn];
    #pragma unroll
    for (int mt = 0; mt < 4; ++mt) {
      #pragma unroll
      for (int vr = 0; vr < 8; ++vr) {
        const long gm = blockM + wm * 64 + mt * 16 + vr + 8 * kh;
        float v = fmaxf(acc[mt][nt][vr] + bb, 0.0f);
        const size_t o = (size_t)gm * N + gn;
        if (mode == 0) {
          outB[o] = (__bf16)v;
        } else {
          const float r = resid[o] + v;
          outF[o] = r;
          outB[o] = (__bf16)r;
        }
      }
    }
  }
}

// ---------------------------------------------------------------------------
// Final projection: out[B x 20] = h[B x 512] @ Wf[512 x 20] + bf  (fp32)
// ---------------------------------------------------------------------------
__global__ __launch_bounds__(256)
void k_final(const float* __restrict__ h, const float* __restrict__ Wf,
             const float* __restrict__ bfv, float* __restrict__ out) {
  const int idx = blockIdx.x * blockDim.x + threadIdx.x;
  if (idx >= BDIM * NOUT) return;
  const int b = idx / NOUT, o = idx % NOUT;
  float acc = bfv[o];
  const float* hrow = h + (size_t)b * HDIM;
  #pragma unroll 8
  for (int k = 0; k < HDIM; ++k) acc += hrow[k] * Wf[k * NOUT + o];
  out[idx] = acc;
}

// ---------------------------------------------------------------------------
extern "C" void kernel_launch(void* const* d_in, const int* in_sizes, int n_in,
                              void* d_out, int out_size, void* d_ws, size_t ws_size,
                              hipStream_t stream) {
  (void)in_sizes; (void)n_in; (void)out_size; (void)ws_size;

  const float* input = (const float*)d_in[0];
  const int*   mask  = (const int*)d_in[1];
  const float* embed = (const float*)d_in[2];
  const float* W1    = (const float*)d_in[3];
  const float* b1    = (const float*)d_in[4];
  const float* W2    = (const float*)d_in[5];
  const float* b2    = (const float*)d_in[6];
  const float* Wf    = (const float*)d_in[7];
  const float* bfv   = (const float*)d_in[8];
  float* out = (float*)d_out;

  char* ws = (char*)d_ws;
  float*  hf  = (float*) (ws);                        //  33,554,432 B
  __bf16* hb  = (__bf16*)(ws +  33554432);            //  16,777,216 B
  __bf16* tb  = (__bf16*)(ws +  50331648);            //  67,108,864 B
  __bf16* w1t = (__bf16*)(ws + 117440512);            //  20,971,520 B
  __bf16* w2t = (__bf16*)(ws + 138412032);            //  20,971,520 B

  // One-shot weight bf16 convert + transpose to N-major (deterministic).
  k_transpose_cvt<<<dim3(HID / 32, HDIM / 32, NBLK), 256, 0, stream>>>(
      W1, w1t, HDIM, HID);
  k_transpose_cvt<<<dim3(HDIM / 32, HID / 32, NBLK), 256, 0, stream>>>(
      W2, w2t, HID, HDIM);

  // Feature pooling -> h (fp32 + bf16).
  k_pool<<<BDIM, DFEAT, 0, stream>>>(input, mask, embed, hf, hb);

  // 10 residual MLP blocks on the matrix cores.
  for (int i = 0; i < NBLK; ++i) {
    k_gemm<<<dim3(HID / 128, BDIM / 128), 256, 0, stream>>>(
        hb, w1t + (size_t)i * HDIM * HID, b1 + (size_t)i * HID,
        nullptr, nullptr, tb, HID, HDIM, 0);
    k_gemm<<<dim3(HDIM / 128, BDIM / 128), 256, 0, stream>>>(
        tb, w2t + (size_t)i * HDIM * HID, b2 + (size_t)i * HDIM,
        hf, hf, hb, HDIM, HID, 1);
  }

  // Final 512 -> 20 projection.
  k_final<<<(BDIM * NOUT + 255) / 256, 256, 0, stream>>>(hf, Wf, bfv, out);
}
